// MMD_VAE_6648609375080
// MI455X (gfx1250) — matrix operand order repaired and checked by reading, not compiled, not run on hardware
//
#include <hip/hip_runtime.h>
#include <hip/hip_bf16.h>
#include <math.h>

typedef unsigned short u16;
typedef unsigned int   u32;
typedef __attribute__((ext_vector_type(16))) __bf16 bf16x16;
typedef __attribute__((ext_vector_type(8)))  float  f32x8;
typedef int v4i __attribute__((vector_size(16)));

#define NTOK   4096   // B*S
#define DMODEL 1024
#define DINNER 4096
#define DLAT   64
#define KCHUNK 128
#define LDS_STRIDE 136   // u16 units: 272B = 17*16B -> conflict-free padded rows

struct __align__(16) U128 { u32 w[4]; };
union BFV { U128 q[2]; bf16x16 v; };

__device__ __forceinline__ u16 f2bf(float f) {
  u32 u = __float_as_uint(f);
  u32 r = u + 0x7FFFu + ((u >> 16) & 1u);   // round-to-nearest-even
  return (u16)(r >> 16);
}
__device__ __forceinline__ float bf2f(u16 h) {
  return __uint_as_float(((u32)h) << 16);
}

// ---------------- async global -> LDS staging (CDNA5 path, with fallback) ----------------
#if __has_builtin(__builtin_amdgcn_global_load_async_to_lds_b128)
#define HAS_ASYNC_LDS 1
__device__ __forceinline__ void async_copy16(const u16* g, u16* l) {
  __builtin_amdgcn_global_load_async_to_lds_b128(
      (__attribute__((address_space(1))) v4i*)g,
      (__attribute__((address_space(3))) v4i*)l, 0, 0);
}
__device__ __forceinline__ void async_wait() {
#if __has_builtin(__builtin_amdgcn_s_wait_asynccnt)
  __builtin_amdgcn_s_wait_asynccnt(0);
#else
  asm volatile("s_wait_asynccnt 0" ::: "memory");
#endif
}
#else
#define HAS_ASYNC_LDS 0
__device__ __forceinline__ void async_copy16(const u16* g, u16* l) {
  *(U128*)l = *(const U128*)g;
}
__device__ __forceinline__ void async_wait() {}
#endif

// A operand: 16x32 bf16, row-major [M][K] source.
// lane l: row m0+(l&15); elems 0..7 at K=k0+8*(l>>4), elems 8..15 at K=k0+16+8*(l>>4)
__device__ __forceinline__ bf16x16 load_a16(const u16* A, int ldk, int m0, int k0, int lane) {
  int m  = m0 + (lane & 15);
  int hf = lane >> 4;
  const u16* p = A + (size_t)m * ldk + k0 + 8 * hf;
  BFV r;
  r.q[0] = *(const U128*)(p);
  r.q[1] = *(const U128*)(p + 16);
  return r.v;
}
// A operand fragment from LDS-staged tile (row stride LDS_STRIDE u16, 32-elem k-steps)
__device__ __forceinline__ bf16x16 load_a_lds(const u16* sa, int mtile, int ks, int lane) {
  int row = mtile + (lane & 15);
  int hf = lane >> 4;
  const u16* p = sa + row * LDS_STRIDE + ks * 32 + 8 * hf;
  BFV r;
  r.q[0] = *(const U128*)(p);
  r.q[1] = *(const U128*)(p + 16);
  return r.v;
}
// B operand: 32x16 bf16 from transposed weight [N][K] row-major.
// lane l: col n0+(l&15); elems 0..7 at K=k0+8*(l>>4), elems 8..15 at K=k0+16+8*(l>>4)
__device__ __forceinline__ bf16x16 load_b16(const u16* BT, int ldk, int n0, int k0, int lane) {
  int n  = n0 + (lane & 15);
  int hf = lane >> 4;
  const u16* p = BT + (size_t)n * ldk + k0 + 8 * hf;
  BFV r;
  r.q[0] = *(const U128*)(p);
  r.q[1] = *(const U128*)(p + 16);
  return r.v;
}
__device__ __forceinline__ f32x8 wmma_bf16(bf16x16 a, bf16x16 b, f32x8 c) {
  return __builtin_amdgcn_wmma_f32_16x16x32_bf16(false, a, false, b, (short)0, c, false, false);
}

// ---------------- conversion kernels ----------------

__global__ void k_cast(const float* __restrict__ in, u16* __restrict__ out, int n) {
  int i = blockIdx.x * blockDim.x + threadIdx.x;
  if (i < n) out[i] = f2bf(in[i]);
}

// f32 [R][C] -> bf16 [C][R]
__global__ void k_transpose(const float* __restrict__ in, u16* __restrict__ out, int R, int C) {
  __shared__ float tile[32][33];
  int c0 = blockIdx.x * 32, r0 = blockIdx.y * 32;
  int tx = threadIdx.x & 31, ty = threadIdx.x >> 5;   // 256 threads = 32x8
  for (int i = ty; i < 32; i += 8)
    tile[i][tx] = in[(size_t)(r0 + i) * C + c0 + tx];
  __syncthreads();
  for (int i = ty; i < 32; i += 8)
    out[(size_t)(c0 + i) * R + r0 + tx] = f2bf(tile[tx][i]);
}

// ---------------- LDS-tiled GEMM: block=128thr(4 waves) -> 64(M) x 128(N) ----------------
// A tile (64 x KCHUNK) staged in LDS via async copies, double-buffered; each wave computes
// a 64x32 slice with 4x2 register blocking. ~375B of L2 traffic per WMMA.
template<bool GELU>
__global__ void k_gemm_lds(const u16* __restrict__ A, const u16* __restrict__ BT,
                           const float* __restrict__ bias,
                           u16* __restrict__ outb, float* __restrict__ outf,
                           int N, int K) {
  __shared__ u16 sA[2][64 * LDS_STRIDE];
  const int m0 = blockIdx.y * 64;
  const int wave = threadIdx.x >> 5, lane = threadIdx.x & 31;
  const int n0 = blockIdx.x * 128 + wave * 32;
  const int hf = lane >> 4, cl = lane & 15;
  const int tid = threadIdx.x;
  const int nkc = K / KCHUNK;

  // stage chunk 0: 64 rows x 128 k = 1024 x 16B transfers, 8 per thread
#pragma unroll
  for (int i = 0; i < 8; i++) {
    int idx = tid + 128 * i;
    int row = idx >> 4, seg = idx & 15;
    async_copy16(A + (size_t)(m0 + row) * K + seg * 8,
                 &sA[0][row * LDS_STRIDE + seg * 8]);
  }
  async_wait();
  __syncthreads();

  f32x8 acc[4][2] = {};
  for (int kc = 0; kc < nkc; kc++) {
    if (kc + 1 < nkc) {
      const int kb = (kc + 1) * KCHUNK;
      u16* dst = sA[(kc + 1) & 1];
#pragma unroll
      for (int i = 0; i < 8; i++) {
        int idx = tid + 128 * i;
        int row = idx >> 4, seg = idx & 15;
        async_copy16(A + (size_t)(m0 + row) * K + kb + seg * 8,
                     &dst[row * LDS_STRIDE + seg * 8]);
      }
    }
    const u16* sa = sA[kc & 1];
#pragma unroll
    for (int ks = 0; ks < KCHUNK / 32; ks++) {
      const int kg = kc * KCHUNK + ks * 32;
      bf16x16 b0 = load_b16(BT, K, n0,      kg, lane);
      bf16x16 b1 = load_b16(BT, K, n0 + 16, kg, lane);
      bf16x16 a0 = load_a_lds(sa, 0,  ks, lane);
      bf16x16 a1 = load_a_lds(sa, 16, ks, lane);
      bf16x16 a2 = load_a_lds(sa, 32, ks, lane);
      bf16x16 a3 = load_a_lds(sa, 48, ks, lane);
      acc[0][0] = wmma_bf16(a0, b0, acc[0][0]);
      acc[0][1] = wmma_bf16(a0, b1, acc[0][1]);
      acc[1][0] = wmma_bf16(a1, b0, acc[1][0]);
      acc[1][1] = wmma_bf16(a1, b1, acc[1][1]);
      acc[2][0] = wmma_bf16(a2, b0, acc[2][0]);
      acc[2][1] = wmma_bf16(a2, b1, acc[2][1]);
      acc[3][0] = wmma_bf16(a3, b0, acc[3][0]);
      acc[3][1] = wmma_bf16(a3, b1, acc[3][1]);
    }
    async_wait();
    __syncthreads();
  }

#pragma unroll
  for (int j = 0; j < 2; j++) {
    const int col = n0 + 16 * j + cl;
    const float bv = bias[col];
#pragma unroll
    for (int i = 0; i < 4; i++) {
#pragma unroll
      for (int v = 0; v < 8; v++) {
        int row = m0 + 16 * i + v + 8 * hf;
        float x = acc[i][j][v] + bv;
        if (GELU) {
          float g = 0.5f * x * (1.0f + erff(x * 0.70710678118654752f));
          outb[(size_t)row * N + col] = f2bf(g);
        } else {
          outf[(size_t)row * N + col] = x;
        }
      }
    }
  }
}

// ---------------- register-only GEMM (for K=64 dec GEMM1): one wave -> 64x32 ----------------
__global__ void k_gemm_gelu(const u16* __restrict__ A, const u16* __restrict__ BT,
                            const float* __restrict__ bias, u16* __restrict__ out,
                            int N, int K) {
  const int n0 = blockIdx.x * 32, m0 = blockIdx.y * 64;
  const int lane = threadIdx.x;
  const int hf = lane >> 4, cl = lane & 15;
  f32x8 acc[4][2] = {};
  for (int k = 0; k < K; k += 32) {
    bf16x16 a0 = load_a16(A, K, m0,      k, lane);
    bf16x16 a1 = load_a16(A, K, m0 + 16, k, lane);
    bf16x16 a2 = load_a16(A, K, m0 + 32, k, lane);
    bf16x16 a3 = load_a16(A, K, m0 + 48, k, lane);
    bf16x16 b0 = load_b16(BT, K, n0,      k, lane);
    bf16x16 b1 = load_b16(BT, K, n0 + 16, k, lane);
    acc[0][0] = wmma_bf16(a0, b0, acc[0][0]);
    acc[0][1] = wmma_bf16(a0, b1, acc[0][1]);
    acc[1][0] = wmma_bf16(a1, b0, acc[1][0]);
    acc[1][1] = wmma_bf16(a1, b1, acc[1][1]);
    acc[2][0] = wmma_bf16(a2, b0, acc[2][0]);
    acc[2][1] = wmma_bf16(a2, b1, acc[2][1]);
    acc[3][0] = wmma_bf16(a3, b0, acc[3][0]);
    acc[3][1] = wmma_bf16(a3, b1, acc[3][1]);
  }
#pragma unroll
  for (int j = 0; j < 2; j++) {
    const int col = n0 + 16 * j + cl;
    const float bv = bias[col];
#pragma unroll
    for (int i = 0; i < 4; i++) {
#pragma unroll
      for (int v = 0; v < 8; v++) {
        int row = m0 + 16 * i + v + 8 * hf;
        float x = acc[i][j][v] + bv;
        float g = 0.5f * x * (1.0f + erff(x * 0.70710678118654752f));
        out[(size_t)row * N + col] = f2bf(g);
      }
    }
  }
}

// ---------------- standalone LayerNorm over DMODEL=1024 cols; one block per row ----------------
__global__ void k_ln_row(const float* __restrict__ in,
                         const float* __restrict__ g, const float* __restrict__ b,
                         float* __restrict__ out) {
  __shared__ float s1[256], s2[256];
  __shared__ float s_mu, s_rs;
  const int row = blockIdx.x;
  const float* p = in + (size_t)row * DMODEL;
  float v[4];
  float s = 0.f, q = 0.f;
#pragma unroll
  for (int i = 0; i < 4; i++) {
    v[i] = p[threadIdx.x + 256 * i];
    s += v[i]; q += v[i] * v[i];
  }
  s1[threadIdx.x] = s; s2[threadIdx.x] = q;
  __syncthreads();
  for (int st = 128; st > 0; st >>= 1) {
    if (threadIdx.x < st) {
      s1[threadIdx.x] += s1[threadIdx.x + st];
      s2[threadIdx.x] += s2[threadIdx.x + st];
    }
    __syncthreads();
  }
  if (threadIdx.x == 0) {
    float mu  = s1[0] * (1.0f / DMODEL);
    float var = s2[0] * (1.0f / DMODEL) - mu * mu;
    s_mu = mu; s_rs = rsqrtf(var + 1e-9f);
  }
  __syncthreads();
  float* o = out + (size_t)row * DMODEL;
#pragma unroll
  for (int i = 0; i < 4; i++) {
    int c = threadIdx.x + 256 * i;
    o[c] = (v[i] - s_mu) * s_rs * g[c] + b[c];
  }
}

// ---------------- enc GEMM2 + bias + LayerNorm(64) -> latent f32 + bf16 ----------------
// block = 128 threads (4 waves); block owns 16 rows x 64 cols; wave w -> cols 16w..16w+15
__global__ void k_enc2_ln(const u16* __restrict__ A, const u16* __restrict__ BT,
                          const float* __restrict__ bias,
                          const float* __restrict__ lng, const float* __restrict__ lnb,
                          float* __restrict__ lat_f32, u16* __restrict__ lat_bf) {
  __shared__ float s_sum[16], s_sq[16], s_mu[16], s_rs[16];
  const int m0 = blockIdx.x * 16;
  const int wave = threadIdx.x >> 5, lane = threadIdx.x & 31;
  const int hf = lane >> 4, cl = lane & 15;
  const int n0 = wave * 16;
  if (threadIdx.x < 16) { s_sum[threadIdx.x] = 0.f; s_sq[threadIdx.x] = 0.f; }
  __syncthreads();

  f32x8 acc = {};
  for (int k = 0; k < DINNER; k += 32) {
    bf16x16 a = load_a16(A, DINNER, m0, k, lane);
    bf16x16 b = load_b16(BT, DINNER, n0, k, lane);
    acc = wmma_bf16(a, b, acc);
  }
  const int col = n0 + cl;
  const float bv = bias[col];
  float s[8], q[8];
#pragma unroll
  for (int v = 0; v < 8; v++) { float h = acc[v] + bv; s[v] = h; q[v] = h * h; }
#pragma unroll
  for (int m = 1; m < 16; m <<= 1) {
#pragma unroll
    for (int v = 0; v < 8; v++) {
      s[v] += __shfl_xor(s[v], m, 32);
      q[v] += __shfl_xor(q[v], m, 32);
    }
  }
  if (cl == 0) {
#pragma unroll
    for (int v = 0; v < 8; v++) {
      atomicAdd(&s_sum[v + 8 * hf], s[v]);
      atomicAdd(&s_sq[v + 8 * hf], q[v]);
    }
  }
  __syncthreads();
  if (threadIdx.x < 16) {
    float mu  = s_sum[threadIdx.x] * (1.0f / DLAT);
    float var = s_sq[threadIdx.x] * (1.0f / DLAT) - mu * mu;
    s_mu[threadIdx.x] = mu;
    s_rs[threadIdx.x] = rsqrtf(var + 1e-9f);
  }
  __syncthreads();
  const float gv = lng[col], bb = lnb[col];
#pragma unroll
  for (int v = 0; v < 8; v++) {
    int r = v + 8 * hf;
    int row = m0 + r;
    float y = (acc[v] + bv - s_mu[r]) * s_rs[r] * gv + bb;
    lat_f32[(size_t)row * DLAT + col] = y;
    lat_bf[(size_t)row * DLAT + col] = f2bf(y);
  }
}

// ---------------- row squared norms (from bf16, consistent with WMMA dots) ----------------
__global__ void k_rownorm(const u16* __restrict__ X, float* __restrict__ out) {
  int r = blockIdx.x * blockDim.x + threadIdx.x;
  if (r >= NTOK) return;
  float s = 0.f;
  for (int k = 0; k < DLAT; k++) { float v = bf2f(X[(size_t)r * DLAT + k]); s += v * v; }
  out[r] = s;
}

// ---------------- Gram tile -> sum of exp(-sq/d^2), per-block partial ----------------
__global__ void k_gram(const u16* __restrict__ X, const u16* __restrict__ Y,
                       const float* __restrict__ x2, const float* __restrict__ y2,
                       float* __restrict__ partial) {
  const int m0 = blockIdx.y * 16, n0 = blockIdx.x * 16;
  const int lane = threadIdx.x;
  const int hf = lane >> 4, cl = lane & 15;
  f32x8 c = {};
  c = wmma_bf16(load_a16(X, DLAT, m0, 0,  lane), load_b16(Y, DLAT, n0, 0,  lane), c);
  c = wmma_bf16(load_a16(X, DLAT, m0, 32, lane), load_b16(Y, DLAT, n0, 32, lane), c);
  const float yv = y2[n0 + cl];
  float s = 0.f;
#pragma unroll
  for (int v = 0; v < 8; v++) {
    int row = m0 + v + 8 * hf;
    float sq = x2[row] + yv - 2.0f * c[v];
    s += expf(sq * (-1.0f / 4096.0f));   // d*d = 64*64
  }
#pragma unroll
  for (int m = 1; m < 32; m <<= 1) s += __shfl_xor(s, m, 32);
  if (lane == 0) partial[(size_t)blockIdx.y * gridDim.x + blockIdx.x] = s;
}

// deterministic fixed-order reduction of n partials -> *out
__global__ void k_reduce(const float* __restrict__ p, int n, float* __restrict__ out) {
  __shared__ float sm[256];
  float s = 0.f;
  for (int i = threadIdx.x; i < n; i += 256) s += p[i];
  sm[threadIdx.x] = s;
  __syncthreads();
  for (int st = 128; st > 0; st >>= 1) {
    if (threadIdx.x < st) sm[threadIdx.x] += sm[threadIdx.x + st];
    __syncthreads();
  }
  if (threadIdx.x == 0) *out = sm[0];
}

__global__ void k_final(const float* __restrict__ acc, float* __restrict__ out) {
  const float inv = 1.0f / ((float)NTOK * (float)NTOK);
  float mmd = (acc[0] + acc[1] - 2.0f * acc[2]) * inv;
  out[0] = mmd * 64.0f;   // /B*S = *512/8
}

extern "C" void kernel_launch(void* const* d_in, const int* in_sizes, int n_in,
                              void* d_out, int out_size, void* d_ws, size_t ws_size,
                              hipStream_t stream) {
  (void)in_sizes; (void)n_in; (void)out_size; (void)ws_size;
  const float* hidden = (const float*)d_in[0];
  const float* truep  = (const float*)d_in[1];
  const float* enc_w1 = (const float*)d_in[2];
  const float* enc_b1 = (const float*)d_in[3];
  const float* enc_w2 = (const float*)d_in[4];
  const float* enc_b2 = (const float*)d_in[5];
  const float* enc_g  = (const float*)d_in[6];
  const float* enc_bt = (const float*)d_in[7];
  const float* dec_w1 = (const float*)d_in[8];
  const float* dec_b1 = (const float*)d_in[9];
  const float* dec_w2 = (const float*)d_in[10];
  const float* dec_b2 = (const float*)d_in[11];
  const float* dec_g  = (const float*)d_in[12];
  const float* dec_bt = (const float*)d_in[13];

  float* recon = (float*)d_out;
  float* lat   = recon + (size_t)NTOK * DMODEL;
  float* reg   = lat + (size_t)NTOK * DLAT;

  char* ws = (char*)d_ws;
  size_t off = 0;
  auto alloc = [&](size_t bytes) -> void* {
    void* p = (void*)(ws + off);
    off += (bytes + 255) & ~(size_t)255;
    return p;
  };
  u16* XBF  = (u16*)alloc((size_t)NTOK * DMODEL * 2);   // hidden bf16
  u16* TBF  = (u16*)alloc((size_t)NTOK * DLAT * 2);     // true_samples bf16
  u16* W1ET = (u16*)alloc((size_t)DINNER * DMODEL * 2); // enc_w1^T [DI][D]
  u16* W2ET = (u16*)alloc((size_t)DLAT * DINNER * 2);   // enc_w2^T [DL][DI]
  u16* W1DT = (u16*)alloc((size_t)DINNER * DLAT * 2);   // dec_w1^T [DI][DL]
  u16* W2DT = (u16*)alloc((size_t)DMODEL * DINNER * 2); // dec_w2^T [D][DI]
  u16* H    = (u16*)alloc((size_t)NTOK * DINNER * 2);   // shared hidden (enc then dec)
  u16* LBF  = (u16*)alloc((size_t)NTOK * DLAT * 2);     // latent bf16
  float* H2F = (float*)alloc((size_t)NTOK * DMODEL * 4);// dec GEMM2 pre-LN f32
  float* X2 = (float*)alloc((size_t)NTOK * 4);
  float* Y2 = (float*)alloc((size_t)NTOK * 4);
  float* P0 = (float*)alloc((size_t)(NTOK/16)*(NTOK/16) * 4);
  float* P1 = (float*)alloc((size_t)(NTOK/16)*(NTOK/16) * 4);
  float* P2 = (float*)alloc((size_t)(NTOK/16)*(NTOK/16) * 4);
  float* ACC = (float*)alloc(3 * 4);

  // casts + weight transposes
  k_cast<<<(NTOK * DMODEL) / 256, 256, 0, stream>>>(hidden, XBF, NTOK * DMODEL);
  k_cast<<<(NTOK * DLAT) / 256, 256, 0, stream>>>(truep, TBF, NTOK * DLAT);
  k_transpose<<<dim3(DINNER / 32, DMODEL / 32), 256, 0, stream>>>(enc_w1, W1ET, DMODEL, DINNER);
  k_transpose<<<dim3(DLAT / 32, DINNER / 32), 256, 0, stream>>>(enc_w2, W2ET, DINNER, DLAT);
  k_transpose<<<dim3(DINNER / 32, DLAT / 32), 256, 0, stream>>>(dec_w1, W1DT, DLAT, DINNER);
  k_transpose<<<dim3(DMODEL / 32, DINNER / 32), 256, 0, stream>>>(dec_w2, W2DT, DINNER, DMODEL);

  // encoder: GEMM1 (LDS-tiled, K=1024) + fused GEMM2+LN
  k_gemm_lds<true><<<dim3(DINNER / 128, NTOK / 64), 128, 0, stream>>>(
      XBF, W1ET, enc_b1, H, nullptr, DINNER, DMODEL);
  k_enc2_ln<<<NTOK / 16, 128, 0, stream>>>(H, W2ET, enc_b2, enc_g, enc_bt, lat, LBF);

  // decoder: GEMM1 (K=64, register-only), GEMM2 (LDS-tiled, K=4096, f32 out), row LN
  k_gemm_gelu<<<dim3(DINNER / 32, NTOK / 64), 32, 0, stream>>>(LBF, W1DT, dec_b1, H, DINNER, DLAT);
  k_gemm_lds<false><<<dim3(DMODEL / 128, NTOK / 64), 128, 0, stream>>>(
      H, W2DT, dec_b2, nullptr, H2F, DMODEL, DINNER);
  k_ln_row<<<NTOK, 256, 0, stream>>>(H2F, dec_g, dec_bt, recon);

  // MMD
  k_rownorm<<<NTOK / 256, 256, 0, stream>>>(TBF, X2);
  k_rownorm<<<NTOK / 256, 256, 0, stream>>>(LBF, Y2);
  dim3 gg(NTOK / 16, NTOK / 16);
  k_gram<<<gg, 32, 0, stream>>>(TBF, TBF, X2, X2, P0);
  k_gram<<<gg, 32, 0, stream>>>(LBF, LBF, Y2, Y2, P1);
  k_gram<<<gg, 32, 0, stream>>>(TBF, LBF, X2, Y2, P2);
  const int np = (NTOK / 16) * (NTOK / 16);
  k_reduce<<<1, 256, 0, stream>>>(P0, np, ACC + 0);
  k_reduce<<<1, 256, 0, stream>>>(P1, np, ACC + 1);
  k_reduce<<<1, 256, 0, stream>>>(P2, np, ACC + 2);
  k_final<<<1, 1, 0, stream>>>(ACC, reg);
}
